// Attention_13262859010282
// MI455X (gfx1250) — matrix operand order repaired and checked by reading, not compiled, not run on hardware
//
#include <hip/hip_runtime.h>
#include <hip/hip_bf16.h>

// Problem constants (from the reference)
#define T_SEQ 2048
#define T_EXT 1024
#define DIM   1024
#define NH    16
#define HDIM  64
#define TKEY  (T_SEQ + T_EXT)   // 3072

typedef __attribute__((ext_vector_type(16))) __bf16         v16bf;
typedef __attribute__((ext_vector_type(8)))  float          v8f;
typedef __attribute__((ext_vector_type(8)))  unsigned short us8;
typedef __attribute__((ext_vector_type(16))) unsigned short us16;

// Exact operand type the async-copy builtin expects (from clang diagnostic):
// 16-byte int vector, global (AS1) source / LDS (AS3) destination.
typedef int b128i __attribute__((vector_size(16)));
typedef __attribute__((address_space(1))) b128i* gb128p;
typedef __attribute__((address_space(3))) b128i* lb128p;

// CDNA5 async Global->LDS copy path (ASYNCcnt-tracked), if the toolchain
// exposes the gfx1250 builtins. Fallback: software-pipelined register staging.
#if defined(__gfx1250__) && __has_builtin(__builtin_amdgcn_global_load_async_to_lds_b128)
#define USE_ASYNC_LDS 1
#else
#define USE_ASYNC_LDS 0
#endif

#if USE_ASYNC_LDS
#if __has_builtin(__builtin_amdgcn_s_wait_asynccnt)
#define WAIT_ASYNC() __builtin_amdgcn_s_wait_asynccnt(0)
#else
#define WAIT_ASYNC() asm volatile("s_wait_asynccnt 0x0" ::: "memory")
#endif
#else
#define WAIT_ASYNC() do {} while (0)
#endif

__device__ __forceinline__ unsigned short f32_to_bf16(float f) {
  unsigned u = __builtin_bit_cast(unsigned, f);
  u += 0x7FFFu + ((u >> 16) & 1u);           // round-to-nearest-even
  return (unsigned short)(u >> 16);
}

// ---------------------------------------------------------------------------
// Generic NT GEMM: C[M,N] = A[M,K] * B[N,K]^T ; A,B bf16 row-major, C fp32.
// Block tile BM x BN, 8 waves, each wave WM x WN (MT x NT WMMA tiles).
// Double-buffered LDS; async Global->LDS when available.
// Requires M % BM == 0, N % BN == 0, K % 32 == 0.
// ---------------------------------------------------------------------------
template <int BM, int BN, int WM, int WN>
__global__ __launch_bounds__(256)
void gemm_nt_bf16_wmma(const unsigned short* __restrict__ A,
                       const unsigned short* __restrict__ B,
                       float* __restrict__ C,
                       int K, int lda, int ldb, int ldc)
{
  constexpr int MT = WM / 16;
  constexpr int NT = WN / 16;
  constexpr int WAVES_M = BM / WM;
  static_assert((BM / WM) * (BN / WN) == 8, "8 waves per block");
  constexpr int ACH = (BM * 4) / 256;        // 16B chunks per thread for A tile
  constexpr int BCH = (BN * 4) / 256;        // 16B chunks per thread for B tile

  __shared__ unsigned short As[2][BM][40];   // 32 K-halves + 8 pad (80B rows)
  __shared__ unsigned short Bs[2][BN][40];

  const int tid  = threadIdx.x;
  const int lane = tid & 31;
  const int wave = tid >> 5;
  const int wm   = (wave % WAVES_M) * WM;
  const int wn   = (wave / WAVES_M) * WN;
  const long bm  = (long)blockIdx.y * BM;
  const long bn  = (long)blockIdx.x * BN;
  const int r16  = lane & 15;
  const int hi   = lane >> 4;                // 0: lanes 0-15, 1: lanes 16-31

  const v8f vzero = {0.f,0.f,0.f,0.f,0.f,0.f,0.f,0.f};
  v8f acc[MT][NT];
#pragma unroll
  for (int i = 0; i < MT; ++i)
#pragma unroll
    for (int j = 0; j < NT; ++j) acc[i][j] = vzero;

  auto stage = [&](int buf, int k0) {
#if USE_ASYNC_LDS
#pragma unroll
    for (int i = 0; i < ACH; ++i) {
      int c = tid + i * 256, row = c >> 2, col = (c & 3) << 3;
      __builtin_amdgcn_global_load_async_to_lds_b128(
          (gb128p)(A + (size_t)(bm + row) * lda + (size_t)(k0 + col)),
          (lb128p)&As[buf][row][col], 0, 0);
    }
#pragma unroll
    for (int i = 0; i < BCH; ++i) {
      int c = tid + i * 256, row = c >> 2, col = (c & 3) << 3;
      __builtin_amdgcn_global_load_async_to_lds_b128(
          (gb128p)(B + (size_t)(bn + row) * ldb + (size_t)(k0 + col)),
          (lb128p)&Bs[buf][row][col], 0, 0);
    }
#else
    uint4 av[ACH], bv[BCH];
#pragma unroll
    for (int i = 0; i < ACH; ++i) {
      int c = tid + i * 256, row = c >> 2, col = (c & 3) << 3;
      av[i] = *(const uint4*)(A + (size_t)(bm + row) * lda + (size_t)(k0 + col));
    }
#pragma unroll
    for (int i = 0; i < BCH; ++i) {
      int c = tid + i * 256, row = c >> 2, col = (c & 3) << 3;
      bv[i] = *(const uint4*)(B + (size_t)(bn + row) * ldb + (size_t)(k0 + col));
    }
#pragma unroll
    for (int i = 0; i < ACH; ++i) {
      int c = tid + i * 256, row = c >> 2, col = (c & 3) << 3;
      *(uint4*)&As[buf][row][col] = av[i];
    }
#pragma unroll
    for (int i = 0; i < BCH; ++i) {
      int c = tid + i * 256, row = c >> 2, col = (c & 3) << 3;
      *(uint4*)&Bs[buf][row][col] = bv[i];
    }
#endif
  };

  auto compute = [&](int buf) {
    // B fragments: lane holds col N=r16; lanes<16 K=0..15, lanes>=16 K=16..31
    v16bf bfrag[NT];
#pragma unroll
    for (int j = 0; j < NT; ++j) {
      const unsigned short* pB = &Bs[buf][wn + j * 16 + r16][hi * 16];
      us8 b0 = *(const us8*)pB;
      us8 b1 = *(const us8*)(pB + 8);
      us16 bvv = __builtin_shufflevector(b0, b1, 0,1,2,3,4,5,6,7,8,9,10,11,12,13,14,15);
      bfrag[j] = __builtin_bit_cast(v16bf, bvv);
    }
    // A fragments: lane row M=r16; lanes<16 K=0..15, lanes>=16 K=16..31
#pragma unroll
    for (int i = 0; i < MT; ++i) {
      const unsigned short* pA = &As[buf][wm + i * 16 + r16][hi * 16];
      us8 a0 = *(const us8*)pA;
      us8 a1 = *(const us8*)(pA + 8);
      us16 avv = __builtin_shufflevector(a0, a1, 0,1,2,3,4,5,6,7,8,9,10,11,12,13,14,15);
      v16bf afrag = __builtin_bit_cast(v16bf, avv);
#pragma unroll
      for (int j = 0; j < NT; ++j)
        acc[i][j] = __builtin_amdgcn_wmma_f32_16x16x32_bf16(
            false, afrag, false, bfrag[j], (short)0, acc[i][j], false, false);
    }
  };

  const int S = K >> 5;
  stage(0, 0);
  WAIT_ASYNC();
  __syncthreads();
  for (int s = 0; s < S; ++s) {
    const int buf = s & 1;
    if (s + 1 < S) stage(buf ^ 1, (s + 1) << 5);   // overlap with compute below
    compute(buf);
    if (s + 1 < S) {
      WAIT_ASYNC();
      __syncthreads();
    }
  }

  // C/D layout: VGPR g, lane l -> M = (l<16 ? g : g+8), N = l%16
#pragma unroll
  for (int i = 0; i < MT; ++i)
#pragma unroll
    for (int j = 0; j < NT; ++j) {
      long rbase = bm + wm + i * 16 + hi * 8;
      long cbase = bn + wn + j * 16 + r16;
#pragma unroll
      for (int g = 0; g < 8; ++g)
        C[(size_t)(rbase + g) * ldc + cbase] = acc[i][j][g];
    }
}

// ---------------------------------------------------------------------------
// Elementwise helpers
// ---------------------------------------------------------------------------
__global__ void cvt_bias_kernel(const float* __restrict__ in,
                                const float* __restrict__ bias,
                                unsigned short* __restrict__ out, int n, int d)
{
  int idx = blockIdx.x * 256 + threadIdx.x;
  if (idx >= n) return;
  float b = bias ? bias[idx % d] : 0.f;
  out[idx] = f32_to_bf16(in[idx] + b);
}

__global__ void transpose_cvt_kernel(const float* __restrict__ W,
                                     unsigned short* __restrict__ WT)
{
  int idx = blockIdx.x * 256 + threadIdx.x;     // DIM*DIM total
  int nrow = idx >> 10;
  int kcol = idx & 1023;
  WT[idx] = f32_to_bf16(W[(size_t)kcol * DIM + nrow]);
}

__global__ void pos_emb_kernel(unsigned short* __restrict__ out)
{
  int idx = blockIdx.x * 256 + threadIdx.x;     // T_SEQ*DIM total
  int t = idx >> 10;
  int j = idx & 1023;
  float pseq = (float)(T_SEQ - 1 - t);
  int jj = (j < 512) ? j : (j - 512);
  // inv_freq = 10000^(-2*jj/DIM) = exp(-jj * ln(10000)/512)
  float inv = __expf(-(float)jj * (9.210340371976184f / 512.0f));
  float a = pseq * inv;
  float v = (j < 512) ? sinf(a) : cosf(a);
  out[idx] = f32_to_bf16(v);
}

// vvT[(h*HDIM+hd)][j] = concat(v, ev)[j][h*HDIM+hd]  (row stride TKEY)
__global__ void build_vvt_kernel(const float* __restrict__ v,
                                 const float* __restrict__ ev,
                                 unsigned short* __restrict__ vvT)
{
  int idx = blockIdx.x * 256 + threadIdx.x;     // DIM*TKEY total
  int j   = idx % TKEY;
  int row = idx / TKEY;                         // == h*HDIM + hd == D column
  float val = (j < T_SEQ) ? v[(size_t)j * DIM + row]
                          : ev[(size_t)(j - T_SEQ) * DIM + row];
  vvT[idx] = f32_to_bf16(val);
}

// ---------------------------------------------------------------------------
// Fused mask + relative-shift + scale + softmax for one head.
// One block per query row t; 3072 keys -> 12 per thread.
// rel_shift(raw)[t,j] = raw[t, T-1 + j - t]  for j <= t (rest masked).
// ---------------------------------------------------------------------------
__device__ __forceinline__ float waveMax(float v) {
#pragma unroll
  for (int o = 16; o > 0; o >>= 1) v = fmaxf(v, __shfl_xor(v, o, 32));
  return v;
}
__device__ __forceinline__ float waveSum(float v) {
#pragma unroll
  for (int o = 16; o > 0; o >>= 1) v += __shfl_xor(v, o, 32);
  return v;
}
__device__ __forceinline__ float blockMax(float v) {
  __shared__ float sm[8];
  v = waveMax(v);
  if ((threadIdx.x & 31) == 0) sm[threadIdx.x >> 5] = v;
  __syncthreads();
  float r = sm[0];
#pragma unroll
  for (int i = 1; i < 8; ++i) r = fmaxf(r, sm[i]);
  __syncthreads();
  return r;
}
__device__ __forceinline__ float blockSum(float v) {
  __shared__ float ss[8];
  v = waveSum(v);
  if ((threadIdx.x & 31) == 0) ss[threadIdx.x >> 5] = v;
  __syncthreads();
  float r = 0.f;
#pragma unroll
  for (int i = 0; i < 8; ++i) r += ss[i];
  __syncthreads();
  return r;
}

__global__ __launch_bounds__(256)
void softmax_row_kernel(const float* __restrict__ content,
                        const float* __restrict__ relraw,
                        const float* __restrict__ eattn,
                        unsigned short* __restrict__ pout)
{
  const int t   = blockIdx.x;
  const int tid = threadIdx.x;
  float sv[12];
#pragma unroll
  for (int i = 0; i < 12; ++i) {
    int j = tid + i * 256;
    float s;
    if (j < T_SEQ) {
      s = (j <= t)
            ? (content[(size_t)t * T_SEQ + j] +
               relraw[(size_t)t * T_SEQ + (T_SEQ - 1 + j - t)])
            : -1e9f;
    } else {
      s = eattn[(size_t)t * T_EXT + (j - T_SEQ)];
    }
    sv[i] = s * 0.125f;                         // 1/sqrt(HDIM)
  }
  float m = sv[0];
#pragma unroll
  for (int i = 1; i < 12; ++i) m = fmaxf(m, sv[i]);
  m = blockMax(m);

  float ex[12];
  float sum = 0.f;
#pragma unroll
  for (int i = 0; i < 12; ++i) { ex[i] = __expf(sv[i] - m); sum += ex[i]; }
  sum = blockSum(sum);
  float inv = 1.0f / sum;
#pragma unroll
  for (int i = 0; i < 12; ++i)
    pout[(size_t)t * TKEY + tid + i * 256] = f32_to_bf16(ex[i] * inv);
}

// ---------------------------------------------------------------------------
// Launcher
// ---------------------------------------------------------------------------
extern "C" void kernel_launch(void* const* d_in, const int* in_sizes, int n_in,
                              void* d_out, int out_size, void* d_ws, size_t ws_size,
                              hipStream_t stream)
{
  (void)in_sizes; (void)n_in; (void)out_size; (void)ws_size;
  const float* x     = (const float*)d_in[0];
  const float* extra = (const float*)d_in[1];
  // d_in[2] = mask (tril, applied analytically), d_in[3] = extra_mask (all ones)
  const float* Wq  = (const float*)d_in[4];
  const float* Wk  = (const float*)d_in[5];
  const float* Wv  = (const float*)d_in[6];
  const float* Wek = (const float*)d_in[7];
  const float* Wev = (const float*)d_in[8];
  const float* Wr  = (const float*)d_in[9];
  const float* Wo  = (const float*)d_in[10];
  const float* rwb = (const float*)d_in[11];   // [NH,HDIM] == [DIM]
  const float* rrb = (const float*)d_in[12];
  float* out = (float*)d_out;

  char* cur = (char*)d_ws;
  auto take = [&](size_t bytes) -> void* {
    void* r = (void*)cur;
    cur += (bytes + 255) & ~(size_t)255;
    return r;
  };

  const size_t TD  = (size_t)T_SEQ * DIM;
  const size_t ED  = (size_t)T_EXT * DIM;
  const size_t DD  = (size_t)DIM * DIM;

  unsigned short* x_bf     = (unsigned short*)take(TD * 2);
  unsigned short* extra_bf = (unsigned short*)take(ED * 2);
  unsigned short* pos_bf   = (unsigned short*)take(TD * 2);
  unsigned short* WqT = (unsigned short*)take(DD * 2);
  unsigned short* WkT = (unsigned short*)take(DD * 2);
  unsigned short* WvT = (unsigned short*)take(DD * 2);
  unsigned short* WekT= (unsigned short*)take(DD * 2);
  unsigned short* WevT= (unsigned short*)take(DD * 2);
  unsigned short* WrT = (unsigned short*)take(DD * 2);
  unsigned short* WoT = (unsigned short*)take(DD * 2);
  float* qf  = (float*)take(TD * 4);
  float* kf  = (float*)take(TD * 4);
  float* vf  = (float*)take(TD * 4);
  float* rf  = (float*)take(TD * 4);
  float* ekf = (float*)take(ED * 4);
  float* evf = (float*)take(ED * 4);
  unsigned short* q_bf  = (unsigned short*)take(TD * 2);
  unsigned short* qw_bf = (unsigned short*)take(TD * 2);
  unsigned short* qr_bf = (unsigned short*)take(TD * 2);
  unsigned short* k_bf  = (unsigned short*)take(TD * 2);
  unsigned short* r_bf  = (unsigned short*)take(TD * 2);
  unsigned short* ek_bf = (unsigned short*)take(ED * 2);
  unsigned short* vvT   = (unsigned short*)take((size_t)DIM * TKEY * 2);
  float* contentB = (float*)take((size_t)T_SEQ * T_SEQ * 4);
  float* relB     = (float*)take((size_t)T_SEQ * T_SEQ * 4);
  float* eattB    = (float*)take((size_t)T_SEQ * T_EXT * 4);
  unsigned short* pB = (unsigned short*)take((size_t)T_SEQ * TKEY * 2);
  float* attn_out = (float*)take(TD * 4);
  unsigned short* attn_bf = (unsigned short*)take(TD * 2);

  // ---- stage 1: conversions / pos emb / weight transposes ----
  cvt_bias_kernel<<<(TD + 255) / 256, 256, 0, stream>>>(x, nullptr, x_bf, (int)TD, DIM);
  cvt_bias_kernel<<<(ED + 255) / 256, 256, 0, stream>>>(extra, nullptr, extra_bf, (int)ED, DIM);
  pos_emb_kernel<<<TD / 256, 256, 0, stream>>>(pos_bf);
  transpose_cvt_kernel<<<DD / 256, 256, 0, stream>>>(Wq,  WqT);
  transpose_cvt_kernel<<<DD / 256, 256, 0, stream>>>(Wk,  WkT);
  transpose_cvt_kernel<<<DD / 256, 256, 0, stream>>>(Wv,  WvT);
  transpose_cvt_kernel<<<DD / 256, 256, 0, stream>>>(Wek, WekT);
  transpose_cvt_kernel<<<DD / 256, 256, 0, stream>>>(Wev, WevT);
  transpose_cvt_kernel<<<DD / 256, 256, 0, stream>>>(Wr,  WrT);
  transpose_cvt_kernel<<<DD / 256, 256, 0, stream>>>(Wo,  WoT);

  // Block tile 128x128 (wave 64x32), and 128x64 (wave 32x32) for N==64.
#define GEMM128(A_, B_, C_, M_, N_, K_, lda_, ldb_, ldc_)                      \
  gemm_nt_bf16_wmma<128,128,64,32>                                             \
      <<<dim3((N_) / 128, (M_) / 128), 256, 0, stream>>>(                      \
          A_, B_, C_, K_, lda_, ldb_, ldc_)
#define GEMM64(A_, B_, C_, M_, N_, K_, lda_, ldb_, ldc_)                       \
  gemm_nt_bf16_wmma<128,64,32,32>                                              \
      <<<dim3((N_) / 64, (M_) / 128), 256, 0, stream>>>(                       \
          A_, B_, C_, K_, lda_, ldb_, ldc_)

  // ---- stage 2: projections ----
  GEMM128(x_bf,     WqT,  qf,  T_SEQ, DIM, DIM, DIM, DIM, DIM);
  GEMM128(x_bf,     WkT,  kf,  T_SEQ, DIM, DIM, DIM, DIM, DIM);
  GEMM128(x_bf,     WvT,  vf,  T_SEQ, DIM, DIM, DIM, DIM, DIM);
  GEMM128(pos_bf,   WrT,  rf,  T_SEQ, DIM, DIM, DIM, DIM, DIM);
  GEMM128(extra_bf, WekT, ekf, T_EXT, DIM, DIM, DIM, DIM, DIM);
  GEMM128(extra_bf, WevT, evf, T_EXT, DIM, DIM, DIM, DIM, DIM);

  // ---- stage 3: bf16 operand prep (fuse rel-pos biases into q copies) ----
  cvt_bias_kernel<<<TD / 256, 256, 0, stream>>>(qf, nullptr, q_bf,  (int)TD, DIM);
  cvt_bias_kernel<<<TD / 256, 256, 0, stream>>>(qf, rwb,     qw_bf, (int)TD, DIM);
  cvt_bias_kernel<<<TD / 256, 256, 0, stream>>>(qf, rrb,     qr_bf, (int)TD, DIM);
  cvt_bias_kernel<<<TD / 256, 256, 0, stream>>>(kf, nullptr, k_bf,  (int)TD, DIM);
  cvt_bias_kernel<<<TD / 256, 256, 0, stream>>>(rf, nullptr, r_bf,  (int)TD, DIM);
  cvt_bias_kernel<<<ED / 256, 256, 0, stream>>>(ekf, nullptr, ek_bf, (int)ED, DIM);
  build_vvt_kernel<<<((size_t)DIM * TKEY) / 256, 256, 0, stream>>>(vf, evf, vvT);

  // ---- stage 4: per-head attention (score buffers reused across heads) ----
  for (int h = 0; h < NH; ++h) {
    const int co = h * HDIM;   // column offset into [*, DIM] tensors
    GEMM128(qw_bf + co, k_bf + co, contentB, T_SEQ, T_SEQ, HDIM, DIM, DIM, T_SEQ);
    GEMM128(qr_bf + co, r_bf + co, relB,     T_SEQ, T_SEQ, HDIM, DIM, DIM, T_SEQ);
    GEMM128(q_bf  + co, ek_bf + co, eattB,   T_SEQ, T_EXT, HDIM, DIM, DIM, T_EXT);
    softmax_row_kernel<<<T_SEQ, 256, 0, stream>>>(contentB, relB, eattB, pB);
    // out[t, h*HDIM + n] = sum_j p[t,j] * vvT[h][n][j]
    GEMM64(pB, vvT + (size_t)co * TKEY, attn_out + co,
           T_SEQ, HDIM, TKEY, TKEY, TKEY, DIM);
  }

  // ---- stage 5: output projection ----
  cvt_bias_kernel<<<TD / 256, 256, 0, stream>>>(attn_out, nullptr, attn_bf, (int)TD, DIM);
  GEMM128(attn_bf, WoT, out, T_SEQ, DIM, DIM, DIM, DIM, DIM);
#undef GEMM128
#undef GEMM64
}
